// SparsePropagation_927712936458
// MI455X (gfx1250) — compile-verified, matrix operand order checked by reading, not run on hardware
//
#include <hip/hip_runtime.h>
#include <hip/hip_bf16.h>

typedef __attribute__((ext_vector_type(16))) __bf16 v16bf;
typedef __attribute__((ext_vector_type(8)))  float  v8f;

#define TILE_I 64
#define TILE_J 64
#define DDIM   256
#define NDIM   4096
#define BATCH  4
#define NTILES (NDIM / TILE_J)
#define WS_NEEDED ((size_t)2 * BATCH * NDIM * DDIM * sizeof(unsigned short)) // 16 MiB

__device__ __forceinline__ unsigned short f2bf(float f) {
    union { float f; unsigned u; } x; x.f = f;
    unsigned r = x.u + 0x7FFFu + ((x.u >> 16) & 1u);   // round-to-nearest-even
    return (unsigned short)(r >> 16);
}
__device__ __forceinline__ float bf2f(unsigned short h) {
    union { unsigned u; float f; } x; x.u = ((unsigned)h) << 16;
    return x.f;
}

// Async DMA: 16 bytes per lane, global -> LDS, tracked by ASYNCcnt.
__device__ __forceinline__ void async_copy_b128(unsigned lds_off, const void* gaddr) {
    asm volatile("global_load_async_to_lds_b128 %0, %1, off"
                 :: "v"(lds_off), "v"(gaddr)
                 : "memory");
}
__device__ __forceinline__ void wait_async0() {
    asm volatile("s_wait_asynccnt 0x0" ::: "memory");
}
__device__ __forceinline__ unsigned lds_addr(const void* p) {
    return (unsigned)(size_t)p;   // low 32 bits of generic ptr = LDS byte offset
}

// Load one 16x32 bf16 fragment (A-layout per CDNA5 ISA 7.12.2) from row-major
// memory (works for LDS or global pointers).
__device__ __forceinline__ v16bf load_frag(const unsigned short* src, int r0,
                                           int pitch, int kb, int lane) {
    int r   = r0 + (lane & 15);
    int sel = (lane >> 4) << 3;            // 0 or 8
    union { v16bf v; uint4 u[2]; } x;
    x.u[0] = *(const uint4*)(src + r * pitch + kb + sel);
    x.u[1] = *(const uint4*)(src + r * pitch + kb + 16 + sel);
    return x.v;
}

// ---------------------------------------------------------------------------
// Prepass: val (f32, [B,N,D]) -> Arm bf16 row-major [B,N,D] and Adm bf16
// d-major [B,D,N]. Padded LDS transpose tile keeps the d-major global writes b128-wide.
// ---------------------------------------------------------------------------
#define TPAD 72   // row pitch in halfwords: 144 B = 9*16 -> every row 16B aligned
__global__ __launch_bounds__(256)
void bf16_prepass(const float* __restrict__ val,
                  unsigned short* __restrict__ Arm,
                  unsigned short* __restrict__ Adm) {
    __shared__ unsigned short T[DDIM * TPAD];   // 36 KB
    const int tid = threadIdx.x;
    const int b   = blockIdx.y;
    const int n0  = blockIdx.x * TILE_I;

    const float* src = val + ((size_t)b * NDIM + n0) * DDIM;
    unsigned short* rm = Arm + ((size_t)b * NDIM + n0) * DDIM;

    for (int i = tid; i < TILE_I * DDIM; i += 256) {
        unsigned short h = f2bf(src[i]);
        rm[i] = h;
        int n = i >> 8, d = i & 255;        // i = n*256 + d
        T[d * TPAD + n] = h;
    }
    __syncthreads();

    unsigned short* dm = Adm + ((size_t)b * DDIM + tid) * NDIM + n0;
    const uint4* s4 = (const uint4*)(T + tid * TPAD);
    uint4* d4 = (uint4*)dm;
#pragma unroll
    for (int k = 0; k < 8; ++k) d4[k] = s4[k];
}

// ---------------------------------------------------------------------------
// Fast path: double-buffered async DMA (global->LDS) overlapped with WMMA.
// V_i A-fragments live in registers for the whole kernel.
// ---------------------------------------------------------------------------
__global__ __launch_bounds__(256)
void sparse_prop_fused_async(const unsigned short* __restrict__ Arm,
                             const unsigned short* __restrict__ Adm,
                             const float* __restrict__ state,
                             float* __restrict__ out_state,
                             float* __restrict__ out_val) {
    __shared__ unsigned short Vj[2][TILE_J * DDIM];   // 2 x 32 KB row-major bf16
    __shared__ unsigned short Vt[2][DDIM * TILE_J];   // 2 x 32 KB d-major bf16
    __shared__ unsigned short E [TILE_I * TILE_J];    // 8 KB edges bf16
    __shared__ float stj[2][TILE_J];
    __shared__ float dsp[256];

    const int tid   = threadIdx.x;
    const int lane  = tid & 31;
    const int w     = tid >> 5;
    const int tiA   = w & 3;
    const int tjB   = (w >> 2) * 2;
    const int dhalf = w >> 2;

    const int b  = blockIdx.y;
    const int i0 = blockIdx.x * TILE_I;
    const float scale = 0.0625f;

    // ---- issue async DMA for tile 0 + load persistent V_i fragments ----
    auto issue_tile = [&](int buf, int j0) {
        const char* g_vj = (const char*)(Arm + ((size_t)b * NDIM + j0) * DDIM);
        const unsigned l_vj = lds_addr(&Vj[buf][0]);
#pragma unroll
        for (int k = 0; k < 8; ++k) {
            int byt = (tid + 256 * k) * 16;               // 2048 x 16B chunks
            async_copy_b128(l_vj + byt, g_vj + byt);
        }
        const char* g_vt = (const char*)(Adm + ((size_t)b * DDIM + tid) * NDIM + j0);
        const unsigned l_vt = lds_addr(&Vt[buf][tid * TILE_J]);
#pragma unroll
        for (int k = 0; k < 8; ++k)
            async_copy_b128(l_vt + k * 16, g_vt + k * 16);
        if (tid < TILE_J)
            stj[buf][tid] = state[(size_t)b * NDIM + j0 + tid];
    };

    issue_tile(0, 0);

    // wave-private V_i fragments: rows [i0 + tiA*16, +16), all 8 k-chunks
    v16bf aVi[8];
    {
        const unsigned short* viRow = Arm + ((size_t)b * NDIM + i0 + tiA * 16) * DDIM;
#pragma unroll
        for (int k8 = 0; k8 < 8; ++k8)
            aVi[k8] = load_frag(viRow, 0, DDIM, k8 * 32, lane);
    }

    v8f dv[8];
#pragma unroll
    for (int s = 0; s < 8; ++s) dv[s] = (v8f){0,0,0,0,0,0,0,0};

    float ds_acc = 0.0f;
    const int ds_row = tid & 63;
    const int ds_q   = tid >> 6;

    wait_async0();
    __syncthreads();          // tile 0 resident in LDS for all waves

    for (int jt = 0; jt < NTILES; ++jt) {
        const int cur = jt & 1;

        // kick DMA for the next tile into the other buffer (overlaps compute)
        if (jt + 1 < NTILES) issue_tile(cur ^ 1, (jt + 1) * TILE_J);

        const unsigned short* vjb = &Vj[cur][0];
        const unsigned short* vtb = &Vt[cur][0];

        // ---- stage A: S = (V_i . V_j^T) * scale -> softsign -> E ----
        v8f s0 = (v8f){0,0,0,0,0,0,0,0};
        v8f s1 = (v8f){0,0,0,0,0,0,0,0};
#pragma unroll
        for (int k8 = 0; k8 < 8; ++k8) {
            int kb = k8 * 32;
            v16bf b0 = load_frag(vjb, (tjB + 0) * 16, DDIM, kb, lane);
            v16bf b1 = load_frag(vjb, (tjB + 1) * 16, DDIM, kb, lane);
            s0 = __builtin_amdgcn_wmma_f32_16x16x32_bf16(false, aVi[k8], false, b0,
                                                         (short)0, s0, false, false);
            s1 = __builtin_amdgcn_wmma_f32_16x16x32_bf16(false, aVi[k8], false, b1,
                                                         (short)0, s1, false, false);
        }
        {
            const int n  = lane & 15;
            const int mh = (lane >> 4) << 3;
#pragma unroll
            for (int r = 0; r < 8; ++r) {
                int row = tiA * 16 + r + mh;
                float e0 = s0[r] * scale; e0 = e0 / (1.0f + fabsf(e0));
                float e1 = s1[r] * scale; e1 = e1 / (1.0f + fabsf(e1));
                E[row * TILE_J + (tjB + 0) * 16 + n] = f2bf(e0);
                E[row * TILE_J + (tjB + 1) * 16 + n] = f2bf(e1);
            }
        }
        __syncthreads();

        // ---- stage B: delta_val += E . V_j ----
        v16bf aE0 = load_frag(E, tiA * 16, TILE_J, 0,  lane);
        v16bf aE1 = load_frag(E, tiA * 16, TILE_J, 32, lane);
#pragma unroll
        for (int s = 0; s < 8; ++s) {
            int d0 = dhalf * 128 + s * 16;
            v16bf bt0 = load_frag(vtb, d0, TILE_J, 0,  lane);
            v16bf bt1 = load_frag(vtb, d0, TILE_J, 32, lane);
            dv[s] = __builtin_amdgcn_wmma_f32_16x16x32_bf16(false, aE0, false, bt0,
                                                            (short)0, dv[s], false, false);
            dv[s] = __builtin_amdgcn_wmma_f32_16x16x32_bf16(false, aE1, false, bt1,
                                                            (short)0, dv[s], false, false);
        }

        // ---- delta_state partials ----
        {
            float p = 0.0f;
#pragma unroll
            for (int c = 0; c < 16; ++c)
                p += bf2f(E[ds_row * TILE_J + ds_q * 16 + c]) * stj[cur][ds_q * 16 + c];
            ds_acc += p;
        }

        wait_async0();        // next tile's DMA complete (no-op on last iter)
        __syncthreads();      // all waves done with E/Vj[cur]/Vt[cur]; next tile visible
    }

    // ---- write delta_val ----
    float* dvo = out_val + ((size_t)b * NDIM + i0) * DDIM;
    {
        const int n  = lane & 15;
        const int mh = (lane >> 4) << 3;
#pragma unroll
        for (int s = 0; s < 8; ++s) {
            int d0 = dhalf * 128 + s * 16;
#pragma unroll
            for (int r = 0; r < 8; ++r) {
                int row = tiA * 16 + r + mh;
                dvo[(size_t)row * DDIM + d0 + n] = dv[s][r];
            }
        }
    }

    // ---- reduce + write delta_state ----
    dsp[ds_q * 64 + ds_row] = ds_acc;
    __syncthreads();
    if (tid < TILE_I) {
        float s = dsp[tid] + dsp[64 + tid] + dsp[128 + tid] + dsp[192 + tid];
        out_state[(size_t)b * NDIM + i0 + tid] = s;
    }
}

// ---------------------------------------------------------------------------
// Fallback (ws too small): fused kernel with in-kernel conversion.
// ---------------------------------------------------------------------------
__global__ __launch_bounds__(256)
void sparse_prop_fused_direct(const float* __restrict__ val,
                              const float* __restrict__ state,
                              float* __restrict__ out_state,
                              float* __restrict__ out_val) {
    __shared__ unsigned short Vi[TILE_I * DDIM];
    __shared__ unsigned short Vj[TILE_J * DDIM];
    __shared__ unsigned short Vt[DDIM * TILE_J];
    __shared__ unsigned short E [TILE_I * TILE_J];
    __shared__ float stj[TILE_J];
    __shared__ float dsp[256];

    const int tid   = threadIdx.x;
    const int lane  = tid & 31;
    const int w     = tid >> 5;
    const int tiA   = w & 3;
    const int tjB   = (w >> 2) * 2;
    const int dhalf = w >> 2;

    const int b  = blockIdx.y;
    const int i0 = blockIdx.x * TILE_I;
    const float scale = 0.0625f;

    const float* vib = val + ((size_t)b * NDIM + i0) * DDIM;
    for (int i = tid; i < TILE_I * DDIM; i += 256)
        Vi[i] = f2bf(vib[i]);

    v8f dv[8];
#pragma unroll
    for (int s = 0; s < 8; ++s) dv[s] = (v8f){0,0,0,0,0,0,0,0};

    float ds_acc = 0.0f;
    const int ds_row = tid & 63;
    const int ds_q   = tid >> 6;

    for (int j0 = 0; j0 < NDIM; j0 += TILE_J) {
        __syncthreads();
        const float* vjb = val + ((size_t)b * NDIM + j0) * DDIM;
        for (int i = tid; i < TILE_J * DDIM; i += 256) {
            unsigned short h = f2bf(vjb[i]);
            Vj[i] = h;
            int r = i >> 8, c = i & 255;
            Vt[c * TILE_J + r] = h;
        }
        if (tid < TILE_J) stj[tid] = state[(size_t)b * NDIM + j0 + tid];
        __syncthreads();

        v8f s0 = (v8f){0,0,0,0,0,0,0,0};
        v8f s1 = (v8f){0,0,0,0,0,0,0,0};
#pragma unroll
        for (int kb = 0; kb < DDIM; kb += 32) {
            v16bf a  = load_frag(Vi, tiA * 16,       DDIM, kb, lane);
            v16bf b0 = load_frag(Vj, (tjB + 0) * 16, DDIM, kb, lane);
            v16bf b1 = load_frag(Vj, (tjB + 1) * 16, DDIM, kb, lane);
            s0 = __builtin_amdgcn_wmma_f32_16x16x32_bf16(false, a, false, b0,
                                                         (short)0, s0, false, false);
            s1 = __builtin_amdgcn_wmma_f32_16x16x32_bf16(false, a, false, b1,
                                                         (short)0, s1, false, false);
        }
        {
            const int n  = lane & 15;
            const int mh = (lane >> 4) << 3;
#pragma unroll
            for (int r = 0; r < 8; ++r) {
                int row = tiA * 16 + r + mh;
                float e0 = s0[r] * scale; e0 = e0 / (1.0f + fabsf(e0));
                float e1 = s1[r] * scale; e1 = e1 / (1.0f + fabsf(e1));
                E[row * TILE_J + (tjB + 0) * 16 + n] = f2bf(e0);
                E[row * TILE_J + (tjB + 1) * 16 + n] = f2bf(e1);
            }
        }
        __syncthreads();

        v16bf aE0 = load_frag(E, tiA * 16, TILE_J, 0,  lane);
        v16bf aE1 = load_frag(E, tiA * 16, TILE_J, 32, lane);
#pragma unroll
        for (int s = 0; s < 8; ++s) {
            int d0 = dhalf * 128 + s * 16;
            v16bf bt0 = load_frag(Vt, d0, TILE_J, 0,  lane);
            v16bf bt1 = load_frag(Vt, d0, TILE_J, 32, lane);
            dv[s] = __builtin_amdgcn_wmma_f32_16x16x32_bf16(false, aE0, false, bt0,
                                                            (short)0, dv[s], false, false);
            dv[s] = __builtin_amdgcn_wmma_f32_16x16x32_bf16(false, aE1, false, bt1,
                                                            (short)0, dv[s], false, false);
        }
        {
            float p = 0.0f;
#pragma unroll
            for (int c = 0; c < 16; ++c)
                p += bf2f(E[ds_row * TILE_J + ds_q * 16 + c]) * stj[ds_q * 16 + c];
            ds_acc += p;
        }
    }

    float* dvo = out_val + ((size_t)b * NDIM + i0) * DDIM;
    {
        const int n  = lane & 15;
        const int mh = (lane >> 4) << 3;
#pragma unroll
        for (int s = 0; s < 8; ++s) {
            int d0 = dhalf * 128 + s * 16;
#pragma unroll
            for (int r = 0; r < 8; ++r) {
                int row = tiA * 16 + r + mh;
                dvo[(size_t)row * DDIM + d0 + n] = dv[s][r];
            }
        }
    }

    __syncthreads();
    dsp[ds_q * 64 + ds_row] = ds_acc;
    __syncthreads();
    if (tid < TILE_I) {
        float s = dsp[tid] + dsp[64 + tid] + dsp[128 + tid] + dsp[192 + tid];
        out_state[(size_t)b * NDIM + i0 + tid] = s;
    }
}

extern "C" void kernel_launch(void* const* d_in, const int* in_sizes, int n_in,
                              void* d_out, int out_size, void* d_ws, size_t ws_size,
                              hipStream_t stream) {
    const float* val   = (const float*)d_in[0];   // [4, 4096, 256] f32
    const float* state = (const float*)d_in[1];   // [4, 4096] f32
    float* out = (float*)d_out;
    float* out_state = out;                        // [4, 4096]
    float* out_val   = out + (size_t)BATCH * NDIM; // [4, 4096, 256]

    dim3 grid(NDIM / TILE_I, BATCH);
    dim3 block(256);

    if (ws_size >= WS_NEEDED) {
        unsigned short* Arm = (unsigned short*)d_ws;              // 8 MiB
        unsigned short* Adm = Arm + (size_t)BATCH * NDIM * DDIM;  // 8 MiB
        bf16_prepass<<<grid, block, 0, stream>>>(val, Arm, Adm);
        sparse_prop_fused_async<<<grid, block, 0, stream>>>(Arm, Adm, state,
                                                            out_state, out_val);
    } else {
        sparse_prop_fused_direct<<<grid, block, 0, stream>>>(val, state,
                                                             out_state, out_val);
    }
}
